// MultiheadAttention_30709016166637
// MI455X (gfx1250) — matrix operand order repaired and checked by reading, not compiled, not run on hardware
//
#include <hip/hip_runtime.h>

// ---------------------------------------------------------------------------
// MI455X (gfx1250) fused MHA: bf16 WMMA GEMMs + flash attention.
// B=2, S=2048, IN_DIM=D_MODEL=1024, H=16, HD=64.
//
// Roofline: ~69 GFLOP of matrix math vs ~100 MB of traffic (weights resident
// in 192 MB L2) -> matrix-core bound. All GEMMs + attention run on
// v_wmma_f32_16x16x32_bf16. Softmax scale is folded into W_qkv/b_qkv Q rows;
// row-sums computed with a ones-matrix WMMA (stats land in slot layout).
// ---------------------------------------------------------------------------

typedef __attribute__((ext_vector_type(16))) __bf16 v16bf;
typedef __attribute__((ext_vector_type(8)))  __bf16 v8bf;
typedef __attribute__((ext_vector_type(8)))  float  v8f;

#define BB      2
#define SS      2048
#define INDIM   1024
#define DMODEL  1024
#define NH      16
#define HD      64
#define QKV_LD  (3 * DMODEL)   // 3072

__device__ __forceinline__ v16bf join8(v8bf lo, v8bf hi) {
  union { v16bf v; v8bf h[2]; } u;
  u.h[0] = lo; u.h[1] = hi;
  return u.v;
}

__device__ __forceinline__ v8f wmma_bf16(v16bf a, v16bf b, v8f c) {
  return __builtin_amdgcn_wmma_f32_16x16x32_bf16(false, a, false, b,
                                                 (short)0, c, false, false);
}

// ---------------------------------------------------------------------------
// fp32 -> bf16 conversions
// ---------------------------------------------------------------------------
__global__ void cvt_f32_bf16(const float* __restrict__ in,
                             __bf16* __restrict__ out, int n) {
  int i = blockIdx.x * 256 + threadIdx.x;
  if (i < n) out[i] = (__bf16)in[i];
}

// W_qkv conversion with 1/sqrt(HD) folded into Q-channel rows
// (row o is a Q channel iff (o % 192) < 64).
__global__ void cvt_wqkv_bf16(const float* __restrict__ in,
                              __bf16* __restrict__ out, int n) {
  int i = blockIdx.x * 256 + threadIdx.x;
  if (i >= n) return;
  const int row = i / INDIM;
  const float s = ((row % (3 * HD)) < HD) ? 0.125f : 1.0f;
  out[i] = (__bf16)(in[i] * s);
}

__global__ void scale_bias_q(const float* __restrict__ in,
                             float* __restrict__ out, int n) {
  int i = blockIdx.x * 256 + threadIdx.x;
  if (i < n) out[i] = in[i] * (((i % (3 * HD)) < HD) ? 0.125f : 1.0f);
}

// ---------------------------------------------------------------------------
// C[M,N] = A[M,K](bf16) @ W[N,K]^T(bf16) + bias[N]
// Block: 256 thr = 8 waves (2 M x 4 N); block tile 64(M) x 256(N);
// wave tile 32 x 64: per K=32 step 2 A-frags x 4 B-frags -> 8 v_wmma,
// 12 16B loads (A reused 4x, B reused 2x in registers).
// ---------------------------------------------------------------------------
template <bool OUT_BF16>
__global__ __launch_bounds__(256)
void gemm_xwT(const __bf16* __restrict__ A, const __bf16* __restrict__ W,
              const float* __restrict__ bias, float* __restrict__ Cf,
              __bf16* __restrict__ Cb, int M, int N, int K) {
  const int lane = threadIdx.x & 31;
  const int wave = threadIdx.x >> 5;
  const int l16  = lane & 15;
  const int hi   = (lane < 16) ? 0 : 1;
  const int m0   = blockIdx.x * 64  + (wave & 1) * 32;
  const int n0   = blockIdx.y * 256 + (wave >> 1) * 64;

  v8f acc[2][4];
#pragma unroll
  for (int nt = 0; nt < 4; ++nt) {
    float bv = bias[n0 + nt * 16 + l16];   // column-invariant across rows
#pragma unroll
    for (int r = 0; r < 8; ++r) { acc[0][nt][r] = bv; acc[1][nt][r] = bv; }
  }

  const __bf16* arow0 = A + (size_t)(m0 + l16) * K;
  const __bf16* arow1 = arow0 + (size_t)16 * K;
  const __bf16* wrow0 = W + (size_t)(n0 + l16) * K;

  for (int kb = 0; kb < K; kb += 32) {
    const int abase = kb + hi * 8;
    v16bf a0 = join8(*(const v8bf*)(arow0 + abase),
                     *(const v8bf*)(arow0 + abase + 16));
    v16bf a1 = join8(*(const v8bf*)(arow1 + abase),
                     *(const v8bf*)(arow1 + abase + 16));
    const int bbase = kb + hi * 8;
#pragma unroll
    for (int nt = 0; nt < 4; ++nt) {
      const __bf16* wp = wrow0 + (size_t)nt * 16 * K + bbase;
      v16bf bfrag = join8(*(const v8bf*)(wp), *(const v8bf*)(wp + 16));
      acc[0][nt] = wmma_bf16(a0, bfrag, acc[0][nt]);
      acc[1][nt] = wmma_bf16(a1, bfrag, acc[1][nt]);
    }
  }

#pragma unroll
  for (int mt = 0; mt < 2; ++mt) {
#pragma unroll
    for (int nt = 0; nt < 4; ++nt) {
      const int col = n0 + nt * 16 + l16;
#pragma unroll
      for (int r = 0; r < 8; ++r) {
        const int row = m0 + mt * 16 + r + hi * 8;  // C: vgpr r -> M=r / r+8
        if (OUT_BF16) Cb[(size_t)row * N + col] = (__bf16)acc[mt][nt][r];
        else          Cf[(size_t)row * N + col] = acc[mt][nt][r];
      }
    }
  }
}

// ---------------------------------------------------------------------------
// Flash attention, one (b, h, 64-query-row) tile per 128-thread block.
// Each wave owns 16 query rows; keys streamed in blocks of 32.
// Scores arrive pre-scaled (scale folded into Q). Row max via shfl_xor
// width-16 butterflies; row sum via one ones-matrix WMMA on bf16 P (lands in
// the same vgpr-slot/row layout as the running stats -> no shuffles).
// ---------------------------------------------------------------------------
__global__ __launch_bounds__(128)
void flash_attn(const __bf16* __restrict__ qkv, __bf16* __restrict__ Ov) {
  __shared__ __bf16 kmat[32][64];      // [key][d]
  __shared__ __bf16 vt[64][32];        // [d][key]
  __shared__ __bf16 pbuf[4][16][32];   // per-wave P staging

  const int lane = threadIdx.x & 31;
  const int wave = threadIdx.x >> 5;
  const int l16  = lane & 15;
  const int hi   = (lane < 16) ? 0 : 1;
  const int q0   = blockIdx.x * 64 + wave * 16;
  const int h    = blockIdx.y;
  const int b    = blockIdx.z;
  const size_t hoff = (size_t)h * (3 * HD);

  // Q fragments (d = 0..31, 32..63), loaded once (already scaled by 1/8)
  const __bf16* qp = qkv + ((size_t)(b * SS) + q0 + l16) * QKV_LD + hoff;
  v16bf qA[2];
  {
    const int base = hi * 8;
    qA[0] = join8(*(const v8bf*)(qp + base),      *(const v8bf*)(qp + base + 16));
    qA[1] = join8(*(const v8bf*)(qp + base + 32), *(const v8bf*)(qp + base + 48));
  }

  v16bf onesB;
#pragma unroll
  for (int e = 0; e < 16; ++e) onesB[e] = (__bf16)1.0f;

  float m_r[8], l_r[8];
  v8f o[4];
#pragma unroll
  for (int r = 0; r < 8; ++r) { m_r[r] = -1e30f; l_r[r] = 0.f; }
#pragma unroll
  for (int dt = 0; dt < 4; ++dt)
#pragma unroll
    for (int r = 0; r < 8; ++r) o[dt][r] = 0.f;

  for (int kb = 0; kb < SS; kb += 32) {
    // cooperative staging: K row-major, V transposed
    for (int e = threadIdx.x; e < 32 * 64; e += 128) {
      const int krow = e >> 6, col = e & 63;
      const __bf16* src = qkv + ((size_t)(b * SS) + kb + krow) * QKV_LD + hoff;
      kmat[krow][col] = src[HD + col];
      vt[col][krow]   = src[2 * HD + col];
    }
    __syncthreads();

    // scores for 32 keys (two 16-key column tiles)
    v8f sc[2];
#pragma unroll
    for (int nt = 0; nt < 2; ++nt) {
      const __bf16* kp = &kmat[nt * 16 + l16][0];
      v16bf b0 = join8(*(const v8bf*)(kp + hi * 8), *(const v8bf*)(kp + hi * 8 + 16));        // d 0..31 half
      v16bf b1 = join8(*(const v8bf*)(kp + 32 + hi * 8), *(const v8bf*)(kp + 48 + hi * 8));   // d 32..63 half
      v8f s;
#pragma unroll
      for (int r = 0; r < 8; ++r) s[r] = 0.f;
      s = wmma_bf16(qA[0], b0, s);
      s = wmma_bf16(qA[1], b1, s);
      sc[nt] = s;
    }

    // running max + exp (row = 16 lanes of one half-wave)
    float alpha[8];
#pragma unroll
    for (int r = 0; r < 8; ++r) {
      float bm = fmaxf(sc[0][r], sc[1][r]);
      bm = fmaxf(bm, __shfl_xor(bm, 1, 16));
      bm = fmaxf(bm, __shfl_xor(bm, 2, 16));
      bm = fmaxf(bm, __shfl_xor(bm, 4, 16));
      bm = fmaxf(bm, __shfl_xor(bm, 8, 16));
      const float mn = fmaxf(m_r[r], bm);
      alpha[r] = __expf(m_r[r] - mn);
      sc[0][r] = __expf(sc[0][r] - mn);
      sc[1][r] = __expf(sc[1][r] - mn);
      m_r[r] = mn;
    }

    // P: C-layout -> A-layout through per-wave LDS slice
#pragma unroll
    for (int r = 0; r < 8; ++r) {
      const int row = r + hi * 8;
      pbuf[wave][row][l16]      = (__bf16)sc[0][r];
      pbuf[wave][row][l16 + 16] = (__bf16)sc[1][r];
    }
    asm volatile("s_wait_dscnt 0x0" ::: "memory"); // same-wave LDS in-order
    v16bf pA;
    {
      const __bf16* pp = &pbuf[wave][l16][0];
      const int base = hi * 8;
      pA = join8(*(const v8bf*)(pp + base), *(const v8bf*)(pp + base + 16));
    }

    // row sums of bf16 P via ones-matrix WMMA (C slot layout == stats layout)
    {
      v8f z;
#pragma unroll
      for (int r = 0; r < 8; ++r) z[r] = 0.f;
      v8f rs = wmma_bf16(pA, onesB, z);
#pragma unroll
      for (int r = 0; r < 8; ++r) l_r[r] = l_r[r] * alpha[r] + rs[r];
    }

    // O = diag(alpha) * O + P @ V
#pragma unroll
    for (int dt = 0; dt < 4; ++dt) {
#pragma unroll
      for (int r = 0; r < 8; ++r) o[dt][r] *= alpha[r];
      const __bf16* vp = &vt[dt * 16 + l16][0];
      v16bf vB = join8(*(const v8bf*)(vp + hi * 8), *(const v8bf*)(vp + hi * 8 + 16));
      o[dt] = wmma_bf16(pA, vB, o[dt]);
    }
    __syncthreads();
  }

  // epilogue: normalize and emit bf16 values[b*S+s, h*64+d] for out-proj GEMM
  float inv[8];
#pragma unroll
  for (int r = 0; r < 8; ++r) inv[r] = 1.f / l_r[r];
#pragma unroll
  for (int dt = 0; dt < 4; ++dt) {
    const int col = h * HD + dt * 16 + l16;
#pragma unroll
    for (int r = 0; r < 8; ++r) {
      const int row = q0 + r + hi * 8;
      Ov[((size_t)(b * SS) + row) * DMODEL + col] = (__bf16)(o[dt][r] * inv[r]);
    }
  }
}

// ---------------------------------------------------------------------------
// Launcher
// ---------------------------------------------------------------------------
extern "C" void kernel_launch(void* const* d_in, const int* in_sizes, int n_in,
                              void* d_out, int out_size, void* d_ws, size_t ws_size,
                              hipStream_t stream) {
  (void)in_sizes; (void)n_in; (void)out_size; (void)ws_size;
  const float* x    = (const float*)d_in[0];
  const float* Wqkv = (const float*)d_in[1];
  const float* bqkv = (const float*)d_in[2];
  const float* Wout = (const float*)d_in[3];
  const float* bout = (const float*)d_in[4];
  float* out = (float*)d_out;

  char* w = (char*)d_ws;
  __bf16* xb     = (__bf16*)w;  w += (size_t)BB * SS * INDIM * 2;        // 8 MB
  __bf16* wqkvb  = (__bf16*)w;  w += (size_t)3 * DMODEL * INDIM * 2;     // 6 MB
  __bf16* woutb  = (__bf16*)w;  w += (size_t)DMODEL * DMODEL * 2;        // 2 MB
  __bf16* qkvb   = (__bf16*)w;  w += (size_t)BB * SS * 3 * DMODEL * 2;   // 24 MB
  __bf16* valb   = (__bf16*)w;  w += (size_t)BB * SS * DMODEL * 2;       // 8 MB
  float*  bqkv_s = (float*)w;                                            // 12 KB

  const int nx  = BB * SS * INDIM;
  const int nwq = 3 * DMODEL * INDIM;
  const int nwo = DMODEL * DMODEL;
  const int nbq = 3 * DMODEL;
  cvt_f32_bf16 <<<(nx  + 255) / 256, 256, 0, stream>>>(x,    xb,    nx);
  cvt_wqkv_bf16<<<(nwq + 255) / 256, 256, 0, stream>>>(Wqkv, wqkvb, nwq);
  cvt_f32_bf16 <<<(nwo + 255) / 256, 256, 0, stream>>>(Wout, woutb, nwo);
  scale_bias_q <<<(nbq + 255) / 256, 256, 0, stream>>>(bqkv, bqkv_s, nbq);

  // qkv = x @ Wqkv^T + bqkv (Q channels pre-scaled by 1/8)
  dim3 g1(BB * SS / 64, (3 * DMODEL) / 256);
  gemm_xwT<true><<<g1, 256, 0, stream>>>(xb, wqkvb, bqkv_s, nullptr, qkvb,
                                         BB * SS, 3 * DMODEL, INDIM);

  // flash attention -> values (bf16)
  dim3 g2(SS / 64, NH, BB);
  flash_attn<<<g2, 128, 0, stream>>>(qkvb, valb);

  // out = values @ Wout^T + bout (fp32)
  dim3 g3(BB * SS / 64, DMODEL / 256);
  gemm_xwT<false><<<g3, 256, 0, stream>>>(valb, woutb, bout, out, nullptr,
                                          BB * SS, DMODEL, INDIM);
}
